// PatchedMixtralAttention_48249662603721
// MI455X (gfx1250) — matrix operand order repaired and checked by reading, not compiled, not run on hardware
//
#include <hip/hip_runtime.h>

// ---------------------------------------------------------------------------
// Mixtral sliding-window GQA attention layer for gfx1250 (MI455X).
// bf16 WMMA (v_wmma_f32_16x16x32_bf16) with fp32 accumulation throughout.
// GEMMs double-buffer LDS tiles via the Tensor Data Mover (tensor_load_to_lds,
// TENSORcnt-tracked), per-wave 16-row slices, TDM-padded rows (144B) so the
// 16 row-parallel fragment loads hit 16 distinct LDS banks.
// ---------------------------------------------------------------------------

#define NBATCH 4
#define SEQ    1024
#define NTOK   (NBATCH * SEQ)     // 4096 tokens
#define NHEAD  32
#define NKVH   8
#define HDIM   128
#define DMODEL 4096
#define NQKV   6144               // 4096 q + 1024 k + 1024 v
#define WIN    512
#define PGBS   64                 // paged-cache block size
#define NBLK   72                 // total cache blocks
#define NSLOT  (NBLK * PGBS)      // 4608 cache slots per kv head
#define QSCALE 0.08838834764831845f   // 1/sqrt(128)
#define LROW   72                 // padded LDS row: 64 bf16 + 8 pad = 144 B

typedef __attribute__((ext_vector_type(16))) __bf16 v16bf;
typedef __attribute__((ext_vector_type(8)))  __bf16 v8bf;
typedef __attribute__((ext_vector_type(8)))  float  v8f;
typedef __attribute__((ext_vector_type(4)))  unsigned int v4u;
typedef __attribute__((ext_vector_type(8)))  unsigned int v8u;

__device__ __forceinline__ v8f vzero8() {
  v8f x;
#pragma unroll
  for (int i = 0; i < 8; ++i) x[i] = 0.0f;
  return x;
}

__device__ __forceinline__ v16bf cat8(v8bf lo, v8bf hi) {
  v16bf r;
#pragma unroll
  for (int e = 0; e < 8; ++e) { r[e] = lo[e]; r[8 + e] = hi[e]; }
  return r;
}

__device__ __forceinline__ v8f wmma_bf16(v16bf a, v16bf b, v8f c) {
  return __builtin_amdgcn_wmma_f32_16x16x32_bf16(false, a, false, b,
                                                 (short)0, c, false, false);
}

// 16-lane butterfly reductions via ds_swizzle (xor masks 1,2,4,8 never cross
// the 16-lane half, so both halves reduce independently).
template <int IMM>
__device__ __forceinline__ float swzf(float v) {
  return __int_as_float(__builtin_amdgcn_ds_swizzle(__float_as_int(v), IMM));
}
__device__ __forceinline__ float red_max16(float v) {
  v = fmaxf(v, swzf<0x041f>(v));
  v = fmaxf(v, swzf<0x081f>(v));
  v = fmaxf(v, swzf<0x101f>(v));
  v = fmaxf(v, swzf<0x201f>(v));
  return v;
}
__device__ __forceinline__ float red_sum16(float v) {
  v += swzf<0x041f>(v);
  v += swzf<0x081f>(v);
  v += swzf<0x101f>(v);
  v += swzf<0x201f>(v);
  return v;
}

// ---------------------------------------------------------------------------
// Tensor Data Mover: DMA a 16-row x 128-byte 2D tile (row stride = K bf16
// elements) from global memory into LDS, with the TDM pad unit inserting
// 16 B after every 128 B so LDS rows land at a 144 B (36-dword) stride:
// conflict-free across the 64 LDS banks for 16 row-parallel fragment loads.
// D# per CDNA5 ISA ch.8:
//  group0 = {count=1, lds_addr, global_addr[56:0], type=2}
//  group1 = {data_size=3 (8B), pad_enable=1, pad_interval=4 (32 dwords),
//            pad_amount=3 (4 dwords), tensor_dim0=16, tensor_dim1=16,
//            tile_dim0=16, tile_dim1=16, tensor_dim0_stride=K/4}
// workgroup_mask=0 (not in a cluster); 2-group form (VADDR2/3 NULL, 2D).
// ---------------------------------------------------------------------------
__device__ __forceinline__ void tdm_tile16(const __bf16* g, unsigned ldsoff,
                                           unsigned k_units) {
  const unsigned long long ga = (unsigned long long)(size_t)g;
  v4u g0;
  g0[0] = 1u;                                   // count=1, user descriptor
  g0[1] = ldsoff;                               // LDS byte address
  g0[2] = (unsigned)ga;                         // global_addr[31:0]
  g0[3] = ((unsigned)(ga >> 32) & 0x01FFFFFFu)  // global_addr[56:32]
          | 0x80000000u;                        // type=2 ("image")
  v8u g1;
  g1[0] = (3u << 16)      // data_size = 3 (8-byte units)
        | (1u << 20)      // pad_enable
        | (4u << 22)      // pad_interval: every 32 dwords (= 128 B row)
        | (3u << 25);     // pad_amount: 4 dwords (= 16 B)
  g1[1] = 16u << 16;      // tensor_dim0[15:0] = 16 units (128 B row)
  g1[2] = 16u << 16;      // tensor_dim1[15:0] = 16 rows
  g1[3] = 16u << 16;      // tile_dim0 = 16 units
  g1[4] = 16u;            // tile_dim1 = 16 rows (tile_dim2 = 0)
  g1[5] = k_units;        // tensor_dim0_stride[31:0] (K/4 8B units)
  g1[6] = 0u;
  g1[7] = 0u;
  asm volatile("tensor_load_to_lds %0, %1" :: "s"(g0), "s"(g1) : "memory");
}

__device__ __forceinline__ unsigned lds_off(const void* p) {
  return (unsigned)(size_t)p;  // low 32 bits of a generic LDS pointer = offset
}

// ---------------------------------------------------------------------------
// fp32 -> bf16 elementwise convert (vectorized float4 loads)
// ---------------------------------------------------------------------------
__global__ __launch_bounds__(256)
void cvt_f32_bf16(const float4* __restrict__ x, __bf16* __restrict__ y, int n4) {
  int i = blockIdx.x * 256 + threadIdx.x;
  if (i >= n4) return;
  float4 v = x[i];
  size_t o = (size_t)i * 4;
  y[o + 0] = (__bf16)v.x;
  y[o + 1] = (__bf16)v.y;
  y[o + 2] = (__bf16)v.z;
  y[o + 3] = (__bf16)v.w;
}

// ---------------------------------------------------------------------------
// W[K][N] fp32 -> Wt[N][K] bf16 (LDS tile transpose, 32x32 tiles)
// ---------------------------------------------------------------------------
__global__ __launch_bounds__(256)
void transpose_cvt(const float* __restrict__ W, __bf16* __restrict__ Wt,
                   int K, int N) {
  __shared__ float tile[32][33];
  const int bn = blockIdx.x * 32;   // n tile
  const int bk = blockIdx.y * 32;   // k tile
  const int tx = threadIdx.x & 31;
  const int ty = threadIdx.x >> 5;  // 0..7
#pragma unroll
  for (int i = 0; i < 32; i += 8)
    tile[ty + i][tx] = W[(size_t)(bk + ty + i) * N + bn + tx];
  __syncthreads();
#pragma unroll
  for (int i = 0; i < 32; i += 8)
    Wt[(size_t)(bn + ty + i) * K + bk + tx] = (__bf16)tile[tx][ty + i];
}

// ---------------------------------------------------------------------------
// C[M][N] = A[M][K](bf16, row-major) x Bt[N][K](bf16, i.e. B pre-transposed)
// Block tile 128x128, 8 waves, each wave 32x64 (2x4 WMMA tiles), K-step 64
// (two 32-deep WMMA sub-steps per stage). Double-buffered LDS staging via
// TDM: each wave DMAs its 16-row slice of the A and B tiles;
// wait s_wait_tensorcnt -> barrier -> issue next stage -> 16 WMMAs.
// ---------------------------------------------------------------------------
template <bool OUT_BF16>
__global__ __launch_bounds__(256)
void gemm_bf16(const __bf16* __restrict__ A, const __bf16* __restrict__ Bt,
               void* __restrict__ C, int M, int N, int K) {
  __shared__ __attribute__((aligned(64))) __bf16 As[2][128][LROW];
  __shared__ __attribute__((aligned(64))) __bf16 Bs[2][128][LROW];

  const int tid  = threadIdx.x;
  const int lane = tid & 31;
  const int wave = tid >> 5;
  const int half = lane >> 4;
  const int l16  = lane & 15;
  const int m0 = blockIdx.y * 128;
  const int n0 = blockIdx.x * 128;
  const int wm = (wave & 3) * 32;   // wave row offset in block tile
  const int wn = (wave >> 2) * 64;  // wave col offset in block tile

  v8f acc[2][4];
#pragma unroll
  for (int i = 0; i < 2; ++i)
#pragma unroll
    for (int j = 0; j < 4; ++j) acc[i][j] = vzero8();

  const unsigned kunits = (unsigned)(K / 4);   // row stride in 8B units
  const __bf16* Arow = A  + (size_t)(m0 + wave * 16) * K;
  const __bf16* Brow = Bt + (size_t)(n0 + wave * 16) * K;

  // prologue: DMA stage 0 into buffer 0 (2 TDM ops per wave)
  tdm_tile16(Arow, lds_off(&As[0][wave * 16][0]), kunits);
  tdm_tile16(Brow, lds_off(&Bs[0][wave * 16][0]), kunits);

  const int nk = K / 64;
  int cur = 0;
  for (int i = 0; i < nk; ++i) {
    __builtin_amdgcn_s_wait_tensorcnt(0);   // my slices of buf[cur] landed
    __syncthreads();                        // all slices landed; buf[cur^1] free
    if (i + 1 < nk) {
      const int k1 = (i + 1) * 64;
      tdm_tile16(Arow + k1, lds_off(&As[cur ^ 1][wave * 16][0]), kunits);
      tdm_tile16(Brow + k1, lds_off(&Bs[cur ^ 1][wave * 16][0]), kunits);
    }

#pragma unroll
    for (int ks = 0; ks < 64; ks += 32) {
      v16bf afrag[2], bfrag[4];
#pragma unroll
      for (int ii = 0; ii < 2; ++ii) {
        const __bf16* ap = &As[cur][wm + ii * 16 + l16][ks];
        afrag[ii] = cat8(*(const v8bf*)(ap + half * 16),
                         *(const v8bf*)(ap + half * 16 + 8));
      }
#pragma unroll
      for (int j = 0; j < 4; ++j) {
        const __bf16* bp = &Bs[cur][wn + j * 16 + l16][ks + half * 16];
        bfrag[j] = cat8(*(const v8bf*)bp, *(const v8bf*)(bp + 8));
      }
#pragma unroll
      for (int ii = 0; ii < 2; ++ii)
#pragma unroll
        for (int j = 0; j < 4; ++j)
          acc[ii][j] = wmma_bf16(afrag[ii], bfrag[j], acc[ii][j]);
    }
    cur ^= 1;
  }

#pragma unroll
  for (int i = 0; i < 2; ++i)
#pragma unroll
    for (int j = 0; j < 4; ++j) {
      const int col = n0 + wn + j * 16 + l16;
#pragma unroll
      for (int r = 0; r < 8; ++r) {
        const int row = m0 + wm + i * 16 + r + half * 8;
        if (OUT_BF16)
          ((__bf16*)C)[(size_t)row * N + col] = (__bf16)acc[i][j][r];
        else
          ((float*)C)[(size_t)row * N + col] = acc[i][j][r];
      }
    }
}

// ---------------------------------------------------------------------------
// RoPE + paged KV scatter (through block_offsets, never touching the input
// caches). Q is pre-scaled by 1/sqrt(HD). V stored transposed [kvh][d][slot].
// grid = (NTOK, 48): head-slots 0..31 = q heads, 32..39 = k heads, 40..47 = v.
// ---------------------------------------------------------------------------
__global__ __launch_bounds__(128)
void rope_scatter(const __bf16* __restrict__ qkv,     // [NTOK][NQKV]
                  const int* __restrict__ pos_ids,    // [B][S]
                  const int* __restrict__ bo,         // [B][S/64]
                  __bf16* __restrict__ qb,            // [B*H][S][HD]
                  __bf16* __restrict__ kc,            // [KVH][NSLOT][HD]
                  __bf16* __restrict__ vct) {         // [KVH][HD][NSLOT]
  const int t  = blockIdx.x;
  const int hs = blockIdx.y;
  const int d  = threadIdx.x;
  const int b  = t >> 10;
  const int s  = t & 1023;

  if (hs >= 40) {  // V: plain convert, transposed paged scatter
    const int h = hs - 40;
    const int slot = bo[b * 16 + (s >> 6)] * PGBS + (s & 63);
    vct[((size_t)h * HDIM + d) * NSLOT + slot] =
        qkv[(size_t)t * NQKV + 5120 + h * HDIM + d];
    return;
  }
  if (d >= 64) return;

  const float pos = (float)pos_ids[b * SEQ + s];
  // inv_freq = 1e6^(-d/64) = exp2(-d * log2(1e6)/64)
  const float fr = pos * exp2f(-(float)d * 0.3114307589f);
  float sn, cs;
  __sincosf(fr, &sn, &cs);

  if (hs < 32) {  // Q head
    const __bf16* src = qkv + (size_t)t * NQKV + hs * HDIM;
    const float x0 = (float)src[d], x1 = (float)src[d + 64];
    __bf16* dst = qb + (((size_t)(b * NHEAD + hs)) * SEQ + s) * HDIM;
    dst[d]      = (__bf16)((x0 * cs - x1 * sn) * QSCALE);
    dst[d + 64] = (__bf16)((x1 * cs + x0 * sn) * QSCALE);
  } else {        // K head -> paged cache
    const int h = hs - 32;
    const int slot = bo[b * 16 + (s >> 6)] * PGBS + (s & 63);
    const __bf16* src = qkv + (size_t)t * NQKV + 4096 + h * HDIM;
    const float x0 = (float)src[d], x1 = (float)src[d + 64];
    __bf16* dst = kc + ((size_t)h * NSLOT + slot) * HDIM;
    dst[d]      = (__bf16)(x0 * cs - x1 * sn);
    dst[d + 64] = (__bf16)(x1 * cs + x0 * sn);
  }
}

// ---------------------------------------------------------------------------
// Flash attention, causal + sliding window 512, GQA 4:1.
// 1 wave per (batch, head, 16-query tile); 32-key steps through the paged
// cache; online softmax; P re-layout D->A through per-wave LDS.
// Interior key blocks (fully inside causal+window region) take a maskless
// fast path -- the mask compare/select code runs only on the 1-2 edge blocks.
// ---------------------------------------------------------------------------
__global__ __launch_bounds__(256)
void attn_kernel(const __bf16* __restrict__ qb,    // [B*H][S][HD] (pre-scaled)
                 const __bf16* __restrict__ kc,    // [KVH][NSLOT][HD]
                 const __bf16* __restrict__ vct,   // [KVH][HD][NSLOT]
                 const int* __restrict__ bo,       // [B][16]
                 __bf16* __restrict__ out) {       // [NTOK][DMODEL]
  __shared__ __attribute__((aligned(64))) __bf16 Pl[8][16][32];

  const int lane = threadIdx.x & 31;
  const int wave = threadIdx.x >> 5;
  const int gw   = blockIdx.x * 8 + wave;
  const int qt   = gw & 63;
  const int h    = (gw >> 6) & 31;
  const int b    = gw >> 11;
  const int kvh  = h >> 2;
  const int q0   = qt * 16;
  const int half = lane >> 4;
  const int l16  = lane & 15;

  // Q fragments (A operand), one per 32-wide contraction chunk of HD=128.
  const __bf16* qrow = qb + (((size_t)(b * NHEAD + h)) * SEQ + q0 + l16) * HDIM;
  v16bf qa[4];
#pragma unroll
  for (int i = 0; i < 4; ++i)
    qa[i] = cat8(*(const v8bf*)(qrow + i * 32 + half * 16),
                 *(const v8bf*)(qrow + i * 32 + half * 16 + 8));

  float mrow[8], lrow[8], alpha[8];
  v8f acc[8];
#pragma unroll
  for (int r = 0; r < 8; ++r) { mrow[r] = -1e30f; lrow[r] = 0.0f; }
#pragma unroll
  for (int t = 0; t < 8; ++t) acc[t] = vzero8();

  int kb0 = q0 - (WIN - 1);
  if (kb0 < 0) kb0 = 0;
  kb0 &= ~31;                       // 32-aligned => chunk stays inside a page
  const int kend = q0 + 16;
  const __bf16* kbase = kc  + (size_t)kvh * NSLOT * HDIM;
  const __bf16* vbase = vct + (size_t)kvh * HDIM * NSLOT;
  const int* bob = bo + b * 16;

  for (int kk = kb0; kk < kend; kk += 32) {
    const int slot = bob[kk >> 6] * PGBS + (kk & 63);

    // prefetch next key chunk (L2-resident; pulls toward the WGP)
    const int kk2 = kk + 32;
    if (kk2 < kend) {
      const int sl2 = bob[kk2 >> 6] * PGBS + (kk2 & 63);
      __builtin_prefetch(kbase + (size_t)(sl2 + lane) * HDIM, 0, 1);
    }

    // ---- S = Q * K^T (two 16-key subtiles, 4 WMMAs each over HD) ----
    v8f s[2];
#pragma unroll
    for (int nt = 0; nt < 2; ++nt) {
      v8f c = vzero8();
      const __bf16* kp = kbase + (size_t)(slot + nt * 16 + l16) * HDIM + half * 16;
#pragma unroll
      for (int i = 0; i < 4; ++i)
        c = wmma_bf16(qa[i], *(const v16bf*)(kp + i * 32), c);
      s[nt] = c;
    }

    // ---- online softmax update, P tile to LDS ----
    const bool fullblk = (kk + 31 <= q0) && (kk > q0 + 15 - WIN);
    if (fullblk) {  // wave-uniform: maskless fast path
#pragma unroll
      for (int r = 0; r < 8; ++r) {
        float vmax = red_max16(fmaxf(s[0][r], s[1][r]));
        const float mnew = fmaxf(mrow[r], vmax);
        const float a  = __expf(mrow[r] - mnew);
        const float p0 = __expf(s[0][r] - mnew);
        const float p1 = __expf(s[1][r] - mnew);
        lrow[r] = lrow[r] * a + red_sum16(p0 + p1);
        mrow[r] = mnew;
        alpha[r] = a;
        Pl[wave][r + half * 8][l16]      = (__bf16)p0;
        Pl[wave][r + half * 8][l16 + 16] = (__bf16)p1;
      }
    } else {        // edge blocks: causal + sliding-window masking
#pragma unroll
      for (int r = 0; r < 8; ++r) {
        const int qi = q0 + r + half * 8;
        const int k0i = kk + l16;
        const int k1i = kk + 16 + l16;
        const bool ok0 = (k0i <= qi) && (k0i > qi - WIN);
        const bool ok1 = (k1i <= qi) && (k1i > qi - WIN);
        if (!ok0) s[0][r] = -1e30f;
        if (!ok1) s[1][r] = -1e30f;
        float vmax = red_max16(fmaxf(s[0][r], s[1][r]));
        const float mnew = fmaxf(mrow[r], vmax);
        const float a  = __expf(mrow[r] - mnew);
        const float p0 = ok0 ? __expf(s[0][r] - mnew) : 0.0f;
        const float p1 = ok1 ? __expf(s[1][r] - mnew) : 0.0f;
        lrow[r] = lrow[r] * a + red_sum16(p0 + p1);
        mrow[r] = mnew;
        alpha[r] = a;
        Pl[wave][r + half * 8][l16]      = (__bf16)p0;
        Pl[wave][r + half * 8][l16 + 16] = (__bf16)p1;
      }
    }
    asm volatile("s_wait_dscnt 0x0" ::: "memory");

    // ---- P in A layout ----
    const __bf16* pr = &Pl[wave][l16][0];
    v16bf pa = cat8(*(const v8bf*)(pr + half * 16),
                    *(const v8bf*)(pr + half * 16 + 8));

    // ---- O += P * V (8 output d-tiles), rescale folded into C operand ----
#pragma unroll
    for (int t = 0; t < 8; ++t) {
      const __bf16* vp = vbase + (size_t)(t * 16 + l16) * NSLOT + slot + half * 16;
      v8f c;
#pragma unroll
      for (int r = 0; r < 8; ++r) c[r] = acc[t][r] * alpha[r];
      acc[t] = wmma_bf16(pa, *(const v16bf*)vp, c);
    }
    asm volatile("s_wait_dscnt 0x0" ::: "memory");
  }

  // ---- epilogue: normalize and store bf16 attention output ----
#pragma unroll
  for (int t = 0; t < 8; ++t) {
    const int col = h * HDIM + t * 16 + l16;
#pragma unroll
    for (int r = 0; r < 8; ++r) {
      const int tok = b * SEQ + q0 + r + half * 8;
      out[(size_t)tok * DMODEL + col] = (__bf16)(acc[t][r] / lrow[r]);
    }
  }
}

// ---------------------------------------------------------------------------
// Host-side orchestration
// ---------------------------------------------------------------------------
extern "C" void kernel_launch(void* const* d_in, const int* in_sizes, int n_in,
                              void* d_out, int out_size, void* d_ws, size_t ws_size,
                              hipStream_t stream) {
  (void)in_sizes; (void)n_in; (void)out_size; (void)ws_size;
  const float* hidden = (const float*)d_in[0];
  const float* Wq     = (const float*)d_in[1];
  const float* Wk     = (const float*)d_in[2];
  const float* Wv     = (const float*)d_in[3];
  const float* Wo     = (const float*)d_in[4];
  // d_in[5], d_in[6] = k_cache/v_cache inputs: never written (no input mutation)
  const int* bo  = (const int*)d_in[7];
  const int* pos = (const int*)d_in[8];
  float* out = (float*)d_out;

  unsigned char* w = (unsigned char*)d_ws;
  auto alloc = [&](size_t bytes) -> void* {
    void* p = (void*)w;
    w += (bytes + 255) & ~(size_t)255;
    return p;
  };
  __bf16* Xb     = (__bf16*)alloc((size_t)NTOK * DMODEL * 2);         // hidden bf16
  __bf16* Wqkvt  = (__bf16*)alloc((size_t)NQKV * DMODEL * 2);         // [6144][4096]
  __bf16* Wot    = (__bf16*)alloc((size_t)DMODEL * DMODEL * 2);       // [4096][4096]
  __bf16* qkvr   = (__bf16*)alloc((size_t)NTOK * NQKV * 2);           // raw qkv
  __bf16* qb     = (__bf16*)alloc((size_t)NBATCH * NHEAD * SEQ * HDIM * 2);
  __bf16* kc     = (__bf16*)alloc((size_t)NKVH * NSLOT * HDIM * 2);   // paged K
  __bf16* vct    = (__bf16*)alloc((size_t)NKVH * HDIM * NSLOT * 2);   // paged V^T
  __bf16* ao     = Xb;  // alias: hidden bf16 dead after GEMM1

  // 1) hidden -> bf16
  cvt_f32_bf16<<<(NTOK * DMODEL / 4 + 255) / 256, 256, 0, stream>>>(
      (const float4*)hidden, Xb, NTOK * DMODEL / 4);

  // 2) weight transpose-converts into fused [N][K] bf16 buffers
  transpose_cvt<<<dim3(DMODEL / 32, DMODEL / 32), 256, 0, stream>>>(
      Wq, Wqkvt, DMODEL, DMODEL);
  transpose_cvt<<<dim3(1024 / 32, DMODEL / 32), 256, 0, stream>>>(
      Wk, Wqkvt + (size_t)4096 * DMODEL, DMODEL, 1024);
  transpose_cvt<<<dim3(1024 / 32, DMODEL / 32), 256, 0, stream>>>(
      Wv, Wqkvt + (size_t)5120 * DMODEL, DMODEL, 1024);
  transpose_cvt<<<dim3(DMODEL / 32, DMODEL / 32), 256, 0, stream>>>(
      Wo, Wot, DMODEL, DMODEL);

  // 3) fused QKV projection: [4096 x 4096] x [4096 x 6144] -> bf16
  gemm_bf16<true><<<dim3(NQKV / 128, NTOK / 128), 256, 0, stream>>>(
      Xb, Wqkvt, (void*)qkvr, NTOK, NQKV, DMODEL);

  // 4) RoPE + paged KV scatter
  rope_scatter<<<dim3(NTOK, 48), 128, 0, stream>>>(qkvr, pos, bo, qb, kc, vct);

  // 5) flash attention (8192 wave-tiles / 8 waves per block)
  attn_kernel<<<NBATCH * NHEAD * (SEQ / 16) / 8, 256, 0, stream>>>(
      qb, kc, vct, bo, ao);

  // 6) output projection -> fp32 d_out
  gemm_bf16<false><<<dim3(DMODEL / 128, NTOK / 128), 256, 0, stream>>>(
      ao, Wot, (void*)out, NTOK, DMODEL, DMODEL);
}
